// PurePyTorchKNN_31250182045869
// MI455X (gfx1250) — matrix operand order repaired and checked by reading, not compiled, not run on hardware
//
#include <hip/hip_runtime.h>
#include <math.h>

// Problem constants (from reference)
#define BATCH   4
#define NPTS    8192
#define DIM     128
#define KNN     8

// Tiling
#define QT      64              // queries per workgroup
#define RT      128             // refs per chunk
#define NCHUNK  (NPTS / RT)     // 64
#define THREADS 256             // 8 wave32
#define BSTR    136             // bf16 row stride (elems): 68 dwords -> 4-dword bank skew per row
#define QSTR    132             // f32 fallback row stride
#define RSTR    132

typedef __attribute__((ext_vector_type(2)))  float  v2f;
typedef __attribute__((ext_vector_type(8)))  float  v8f;
typedef __attribute__((ext_vector_type(16))) __bf16 v16bf;

// ---------- bf16 split helpers (round-to-nearest-even) ----------
__device__ __forceinline__ unsigned int f2bf_bits(float x) {
    unsigned int u = __float_as_uint(x);
    return (u + 0x7FFFu + ((u >> 16) & 1u)) >> 16;
}
__device__ __forceinline__ float bfbits2f(unsigned int h) {
    return __uint_as_float(h << 16);
}

__device__ __forceinline__ void insert8(float key, int idx, float (&d)[KNN], int (&ix)[KNN]) {
    if (key < d[KNN - 1]) {
        d[KNN - 1] = key; ix[KNN - 1] = idx;
        #pragma unroll
        for (int j = KNN - 1; j > 0; --j) {
            if (d[j] < d[j - 1]) {
                float td = d[j]; d[j] = d[j - 1]; d[j - 1] = td;
                int   ti = ix[j]; ix[j] = ix[j - 1]; ix[j - 1] = ti;
            }
        }
    }
}

// Frag loader: two b128 LDS reads into one 32B bf16 fragment
__device__ __forceinline__ v16bf ld_frag2(const unsigned short* p0, const unsigned short* p1) {
    union { v16bf v; uint4 u[2]; } f;
    f.u[0] = *(const uint4*)p0;
    f.u[1] = *(const uint4*)p1;
    return f.v;
}

// =====================================================================
// Prep kernel: split f32 -> bf16 hi/lo planes, compute exact row norms.
// One wave32 per row (128 elements => 4 per lane).
// =====================================================================
__global__ __launch_bounds__(THREADS)
void knn_prep_kernel(const float* __restrict__ refp, const float* __restrict__ queryp,
                     unsigned short* __restrict__ refh, unsigned short* __restrict__ refl,
                     unsigned short* __restrict__ qh,   unsigned short* __restrict__ ql,
                     float* __restrict__ ref2, float* __restrict__ q2)
{
    const int wid  = (int)((blockIdx.x * blockDim.x + threadIdx.x) >> 5); // global wave = row id
    const int lane = threadIdx.x & 31;
    const bool isRef = wid < BATCH * NPTS;
    const int row = isRef ? wid : wid - BATCH * NPTS;

    const float* src = (isRef ? refp : queryp) + (size_t)row * DIM + 4 * lane;
    float4 v = *(const float4*)src;

    unsigned int h0 = f2bf_bits(v.x), h1 = f2bf_bits(v.y), h2 = f2bf_bits(v.z), h3 = f2bf_bits(v.w);
    unsigned int l0 = f2bf_bits(v.x - bfbits2f(h0));
    unsigned int l1 = f2bf_bits(v.y - bfbits2f(h1));
    unsigned int l2 = f2bf_bits(v.z - bfbits2f(h2));
    unsigned int l3 = f2bf_bits(v.w - bfbits2f(h3));

    unsigned short* dh = (isRef ? refh : qh) + (size_t)row * DIM + 4 * lane;
    unsigned short* dl = (isRef ? refl : ql) + (size_t)row * DIM + 4 * lane;
    *(uint2*)dh = make_uint2(h0 | (h1 << 16), h2 | (h3 << 16));
    *(uint2*)dl = make_uint2(l0 | (l1 << 16), l2 | (l3 << 16));

    float sq = v.x * v.x + v.y * v.y + v.z * v.z + v.w * v.w;
    #pragma unroll
    for (int off = 16; off >= 1; off >>= 1) sq += __shfl_xor(sq, off);
    if (lane == 0) (isRef ? ref2 : q2)[row] = sq;
}

// =====================================================================
// Main kernel: bf16x3 split-precision GEMM on v_wmma_f32_16x16x32_bf16
// C[M=ref][N=query] so each lane's 8 C VGPRs belong to ONE query.
// __launch_bounds__(256,4): cap at 256 VGPRs -> 4 waves/SIMD -> 2 WGs/WGP.
// =====================================================================
__global__ __launch_bounds__(THREADS, 4)
void knn_wmma_bf16_kernel(const unsigned short* __restrict__ refh,
                          const unsigned short* __restrict__ refl,
                          const unsigned short* __restrict__ qhg,
                          const unsigned short* __restrict__ qlg,
                          const float* __restrict__ ref2,
                          const float* __restrict__ q2g,
                          float* __restrict__ out_val,
                          int*   __restrict__ out_idx)
{
    __shared__ __align__(16) unsigned short sQh[QT * BSTR];   // 17408 B (merge overlay at end)
    __shared__ __align__(16) unsigned short sQl[QT * BSTR];   // 17408 B
    __shared__ __align__(16) unsigned short sRh[RT * BSTR];   // 34816 B
    __shared__ __align__(16) unsigned short sRl[RT * BSTR];   // 34816 B
    __shared__ float sR2[RT];
    __shared__ float sQ2[QT];

    const int tid  = threadIdx.x;
    const int lane = tid & 31;
    const int wave = tid >> 5;
    const int lrow = lane & 15;
    const int half = lane >> 4;

    const int b     = blockIdx.x / (NPTS / QT);
    const int qbase = (blockIdx.x % (NPTS / QT)) * QT;

    // ---- load query tiles (hi & lo planes) ----
    {
        const int qrow  = tid >> 2;      // 0..63
        const int quart = tid & 3;       // 32 elems each
        const size_t g = ((size_t)b * NPTS + qbase + qrow) * DIM + quart * 32;
        const uint4* sh = (const uint4*)(qhg + g);
        const uint4* sl = (const uint4*)(qlg + g);
        uint4* dh = (uint4*)(&sQh[qrow * BSTR + quart * 32]);
        uint4* dl = (uint4*)(&sQl[qrow * BSTR + quart * 32]);
        #pragma unroll
        for (int i = 0; i < 4; ++i) { dh[i] = sh[i]; dl[i] = sl[i]; }
    }
    if (tid < QT) sQ2[tid] = q2g[(size_t)b * NPTS + qbase + tid];
    __syncthreads();

    // per-lane top-K over key = r^2 - 2*q.r
    float bestd[KNN];
    int   besti[KNN];
    #pragma unroll
    for (int i = 0; i < KNN; ++i) { bestd[i] = 3.0e38f; besti[i] = 0; }

    const int qtile = wave >> 1;     // 0..3 : 16-query column tile owned by this wave
    const int rhalf = wave & 1;      // 0..1 : which 64-ref half of the chunk
    const int qo    = qtile * 16;
    const int r0    = rhalf * 64;
    const int qrowN = qo + lrow;     // this lane's query (C-matrix N index)

    for (int c = 0; c < NCHUNK; ++c) {
        __syncthreads();  // previous selection done -> sR*/sR2 reusable

        // ---- load ref chunk (hi & lo planes) + r^2 ----
        {
            const int j   = tid >> 1;    // 0..127
            const int h64 = tid & 1;     // 64 elems each
            const size_t g = ((size_t)b * NPTS + c * RT + j) * DIM + h64 * 64;
            const uint4* sh = (const uint4*)(refh + g);
            const uint4* sl = (const uint4*)(refl + g);
            uint4* dh = (uint4*)(&sRh[j * BSTR + h64 * 64]);
            uint4* dl = (uint4*)(&sRl[j * BSTR + h64 * 64]);
            #pragma unroll
            for (int i = 0; i < 8; ++i) { dh[i] = sh[i]; dl[i] = sl[i]; }
        }
        if (tid < RT) sR2[tid] = ref2[(size_t)b * NPTS + c * RT + tid];
        __syncthreads();

        // ---- WMMA bf16 16x16x32, 3-term compensated ----
        v8f acc0 = {}, acc1 = {}, acc2 = {}, acc3 = {};
        #pragma unroll 2
        for (int k = 0; k < DIM / 32; ++k) {
            // B operand (K x N = 32x16): lane = K in [8h,8h+8) u [16+8h,16+8h+8)
            const unsigned short* qbp = &sQh[qrowN * BSTR + 32 * k + 8 * half];
            const unsigned short* qlp = &sQl[qrowN * BSTR + 32 * k + 8 * half];
            v16bf bh = ld_frag2(qbp, qbp + 16);
            v16bf bl = ld_frag2(qlp, qlp + 16);
            // A operand (M x K = 16x32): lane = K in [8h,8h+8) u [16+8h,16+8h+8)
            {
                const unsigned short* ph = &sRh[(r0 +  0 + lrow) * BSTR + 32 * k + 8 * half];
                const unsigned short* pl = &sRl[(r0 +  0 + lrow) * BSTR + 32 * k + 8 * half];
                v16bf ah = ld_frag2(ph, ph + 16);
                v16bf al = ld_frag2(pl, pl + 16);
                acc0 = __builtin_amdgcn_wmma_f32_16x16x32_bf16(false, ah, false, bh, (short)0, acc0, false, false);
                acc0 = __builtin_amdgcn_wmma_f32_16x16x32_bf16(false, ah, false, bl, (short)0, acc0, false, false);
                acc0 = __builtin_amdgcn_wmma_f32_16x16x32_bf16(false, al, false, bh, (short)0, acc0, false, false);
            }
            {
                const unsigned short* ph = &sRh[(r0 + 16 + lrow) * BSTR + 32 * k + 8 * half];
                const unsigned short* pl = &sRl[(r0 + 16 + lrow) * BSTR + 32 * k + 8 * half];
                v16bf ah = ld_frag2(ph, ph + 16);
                v16bf al = ld_frag2(pl, pl + 16);
                acc1 = __builtin_amdgcn_wmma_f32_16x16x32_bf16(false, ah, false, bh, (short)0, acc1, false, false);
                acc1 = __builtin_amdgcn_wmma_f32_16x16x32_bf16(false, ah, false, bl, (short)0, acc1, false, false);
                acc1 = __builtin_amdgcn_wmma_f32_16x16x32_bf16(false, al, false, bh, (short)0, acc1, false, false);
            }
            {
                const unsigned short* ph = &sRh[(r0 + 32 + lrow) * BSTR + 32 * k + 8 * half];
                const unsigned short* pl = &sRl[(r0 + 32 + lrow) * BSTR + 32 * k + 8 * half];
                v16bf ah = ld_frag2(ph, ph + 16);
                v16bf al = ld_frag2(pl, pl + 16);
                acc2 = __builtin_amdgcn_wmma_f32_16x16x32_bf16(false, ah, false, bh, (short)0, acc2, false, false);
                acc2 = __builtin_amdgcn_wmma_f32_16x16x32_bf16(false, ah, false, bl, (short)0, acc2, false, false);
                acc2 = __builtin_amdgcn_wmma_f32_16x16x32_bf16(false, al, false, bh, (short)0, acc2, false, false);
            }
            {
                const unsigned short* ph = &sRh[(r0 + 48 + lrow) * BSTR + 32 * k + 8 * half];
                const unsigned short* pl = &sRl[(r0 + 48 + lrow) * BSTR + 32 * k + 8 * half];
                v16bf ah = ld_frag2(ph, ph + 16);
                v16bf al = ld_frag2(pl, pl + 16);
                acc3 = __builtin_amdgcn_wmma_f32_16x16x32_bf16(false, ah, false, bh, (short)0, acc3, false, false);
                acc3 = __builtin_amdgcn_wmma_f32_16x16x32_bf16(false, ah, false, bl, (short)0, acc3, false, false);
                acc3 = __builtin_amdgcn_wmma_f32_16x16x32_bf16(false, al, false, bh, (short)0, acc3, false, false);
            }
        }

        // ---- selection: lane owns query qrowN; C VGPR v => ref r0+rt+v+8*half ----
        {
            const int cbase = c * RT;
            #pragma unroll
            for (int v = 0; v < 8; ++v) {
                int lr = r0 + 0 + v + 8 * half;
                insert8(sR2[lr] - 2.0f * acc0[v], cbase + lr, bestd, besti);
            }
            #pragma unroll
            for (int v = 0; v < 8; ++v) {
                int lr = r0 + 16 + v + 8 * half;
                insert8(sR2[lr] - 2.0f * acc1[v], cbase + lr, bestd, besti);
            }
            #pragma unroll
            for (int v = 0; v < 8; ++v) {
                int lr = r0 + 32 + v + 8 * half;
                insert8(sR2[lr] - 2.0f * acc2[v], cbase + lr, bestd, besti);
            }
            #pragma unroll
            for (int v = 0; v < 8; ++v) {
                int lr = r0 + 48 + v + 8 * half;
                insert8(sR2[lr] - 2.0f * acc3[v], cbase + lr, bestd, besti);
            }
        }
    }

    // ---- merge 4 partial sorted lists per query, write out ----
    float* mD = (float*)sQh;                 // overlay: 8192 B
    int*   mI = (int*)(mD + THREADS * KNN);  // next 8192 B (fits in sQh+sQl region)
    #pragma unroll
    for (int i = 0; i < KNN; ++i) { mD[tid * KNN + i] = bestd[i]; mI[tid * KNN + i] = besti[i]; }
    __syncthreads();

    if (tid < QT) {
        const int qt = tid >> 4, lr = tid & 15;
        const int s0 = (2 * qt + 0) * 32 + lr;
        const int s1 = (2 * qt + 0) * 32 + 16 + lr;
        const int s2 = (2 * qt + 1) * 32 + lr;
        const int s3 = (2 * qt + 1) * 32 + 16 + lr;
        int p0 = 0, p1 = 0, p2 = 0, p3 = 0;
        const float q2 = sQ2[tid];
        const size_t obase = ((size_t)b * NPTS + qbase + tid) * KNN;
        for (int k = 0; k < KNN; ++k) {
            float d0 = (p0 < KNN) ? mD[s0 * KNN + p0] : 3.0e38f;
            float d1 = (p1 < KNN) ? mD[s1 * KNN + p1] : 3.0e38f;
            float d2 = (p2 < KNN) ? mD[s2 * KNN + p2] : 3.0e38f;
            float d3 = (p3 < KNN) ? mD[s3 * KNN + p3] : 3.0e38f;
            float dm = d0; int sel = 0;
            if (d1 < dm) { dm = d1; sel = 1; }
            if (d2 < dm) { dm = d2; sel = 2; }
            if (d3 < dm) { dm = d3; sel = 3; }
            int im;
            if      (sel == 0) { im = mI[s0 * KNN + p0]; ++p0; }
            else if (sel == 1) { im = mI[s1 * KNN + p1]; ++p1; }
            else if (sel == 2) { im = mI[s2 * KNN + p2]; ++p2; }
            else               { im = mI[s3 * KNN + p3]; ++p3; }
            out_val[obase + k] = sqrtf(fmaxf(q2 + dm, 0.0f));
            out_idx[obase + k] = im;
        }
    }
}

// =====================================================================
// Fallback: exact-fp32 WMMA kernel (used if workspace too small)
// =====================================================================
__global__ __launch_bounds__(THREADS)
void knn_wmma_f32_kernel(const float* __restrict__ refp,
                         const float* __restrict__ queryp,
                         float* __restrict__ out_val,
                         int*   __restrict__ out_idx)
{
    __shared__ __align__(16) float sQ[QT * QSTR];
    __shared__ __align__(16) float sR[RT * RSTR];
    __shared__ float sPart[THREADS];
    __shared__ float sQ2[QT];
    __shared__ float sR2[RT];

    const int tid  = threadIdx.x;
    const int lane = tid & 31;
    const int wave = tid >> 5;
    const int lrow = lane & 15;
    const int half = lane >> 4;

    const int b     = blockIdx.x / (NPTS / QT);
    const int qbase = (blockIdx.x % (NPTS / QT)) * QT;

    const float* Qg = queryp + ((size_t)b * NPTS + qbase) * DIM;
    const float* Rg = refp   + (size_t)b * NPTS * DIM;

    {
        const int qrow = tid >> 2;
        const int quart = tid & 3;
        const float4* src = (const float4*)(Qg + (size_t)qrow * DIM + quart * 32);
        float4* dst = (float4*)(&sQ[qrow * QSTR + quart * 32]);
        float acc = 0.f;
        #pragma unroll
        for (int i = 0; i < 8; ++i) {
            float4 v = src[i]; dst[i] = v;
            acc += v.x * v.x + v.y * v.y + v.z * v.z + v.w * v.w;
        }
        sPart[tid] = acc;
    }
    __syncthreads();
    if (tid < QT)
        sQ2[tid] = sPart[4 * tid] + sPart[4 * tid + 1] + sPart[4 * tid + 2] + sPart[4 * tid + 3];

    float bestd[KNN];
    int   besti[KNN];
    #pragma unroll
    for (int i = 0; i < KNN; ++i) { bestd[i] = 3.0e38f; besti[i] = 0; }

    const int qtile = wave >> 1;
    const int rhalf = wave & 1;
    const int qo    = qtile * 16;
    const int r0    = rhalf * 64;

    for (int c = 0; c < NCHUNK; ++c) {
        __syncthreads();
        {
            const int j   = tid >> 1;
            const int h64 = tid & 1;
            const float4* src = (const float4*)(Rg + (size_t)(c * RT + j) * DIM + h64 * 64);
            float4* dst = (float4*)(&sR[j * RSTR + h64 * 64]);
            float acc = 0.f;
            #pragma unroll
            for (int i = 0; i < 16; ++i) {
                float4 v = src[i]; dst[i] = v;
                acc += v.x * v.x + v.y * v.y + v.z * v.z + v.w * v.w;
            }
            sPart[tid] = acc;
        }
        __syncthreads();
        if (tid < RT) sR2[tid] = sPart[2 * tid] + sPart[2 * tid + 1];

        v8f acc0 = {}, acc1 = {}, acc2 = {}, acc3 = {};
        {
            const float* qb  = &sQ[(qo + lrow) * QSTR + 2 * half];
            const float* rb0 = &sR[(r0 +  0 + lrow) * RSTR + 2 * half];
            const float* rb1 = &sR[(r0 + 16 + lrow) * RSTR + 2 * half];
            const float* rb2 = &sR[(r0 + 32 + lrow) * RSTR + 2 * half];
            const float* rb3 = &sR[(r0 + 48 + lrow) * RSTR + 2 * half];
            #pragma unroll 8
            for (int k = 0; k < DIM / 4; ++k) {
                v2f bq = *(const v2f*)(qb + 4 * k);
                v2f a0 = *(const v2f*)(rb0 + 4 * k);
                acc0 = __builtin_amdgcn_wmma_f32_16x16x4_f32(false, a0, false, bq, (short)0, acc0, false, false);
                v2f a1 = *(const v2f*)(rb1 + 4 * k);
                acc1 = __builtin_amdgcn_wmma_f32_16x16x4_f32(false, a1, false, bq, (short)0, acc1, false, false);
                v2f a2 = *(const v2f*)(rb2 + 4 * k);
                acc2 = __builtin_amdgcn_wmma_f32_16x16x4_f32(false, a2, false, bq, (short)0, acc2, false, false);
                v2f a3 = *(const v2f*)(rb3 + 4 * k);
                acc3 = __builtin_amdgcn_wmma_f32_16x16x4_f32(false, a3, false, bq, (short)0, acc3, false, false);
            }
        }
        __syncthreads();

        {
            const int cbase = c * RT;
            #pragma unroll
            for (int v = 0; v < 8; ++v) {
                int lr = r0 + 0 + v + 8 * half;
                insert8(sR2[lr] - 2.0f * acc0[v], cbase + lr, bestd, besti);
            }
            #pragma unroll
            for (int v = 0; v < 8; ++v) {
                int lr = r0 + 16 + v + 8 * half;
                insert8(sR2[lr] - 2.0f * acc1[v], cbase + lr, bestd, besti);
            }
            #pragma unroll
            for (int v = 0; v < 8; ++v) {
                int lr = r0 + 32 + v + 8 * half;
                insert8(sR2[lr] - 2.0f * acc2[v], cbase + lr, bestd, besti);
            }
            #pragma unroll
            for (int v = 0; v < 8; ++v) {
                int lr = r0 + 48 + v + 8 * half;
                insert8(sR2[lr] - 2.0f * acc3[v], cbase + lr, bestd, besti);
            }
        }
    }

    float* mD = sQ;
    int*   mI = (int*)(sQ + THREADS * KNN);
    #pragma unroll
    for (int i = 0; i < KNN; ++i) { mD[tid * KNN + i] = bestd[i]; mI[tid * KNN + i] = besti[i]; }
    __syncthreads();

    if (tid < QT) {
        const int qt = tid >> 4, lr = tid & 15;
        const int s0 = (2 * qt + 0) * 32 + lr;
        const int s1 = (2 * qt + 0) * 32 + 16 + lr;
        const int s2 = (2 * qt + 1) * 32 + lr;
        const int s3 = (2 * qt + 1) * 32 + 16 + lr;
        int p0 = 0, p1 = 0, p2 = 0, p3 = 0;
        const float q2 = sQ2[tid];
        const size_t obase = ((size_t)b * NPTS + qbase + tid) * KNN;
        for (int k = 0; k < KNN; ++k) {
            float d0 = (p0 < KNN) ? mD[s0 * KNN + p0] : 3.0e38f;
            float d1 = (p1 < KNN) ? mD[s1 * KNN + p1] : 3.0e38f;
            float d2 = (p2 < KNN) ? mD[s2 * KNN + p2] : 3.0e38f;
            float d3 = (p3 < KNN) ? mD[s3 * KNN + p3] : 3.0e38f;
            float dm = d0; int sel = 0;
            if (d1 < dm) { dm = d1; sel = 1; }
            if (d2 < dm) { dm = d2; sel = 2; }
            if (d3 < dm) { dm = d3; sel = 3; }
            int im;
            if      (sel == 0) { im = mI[s0 * KNN + p0]; ++p0; }
            else if (sel == 1) { im = mI[s1 * KNN + p1]; ++p1; }
            else if (sel == 2) { im = mI[s2 * KNN + p2]; ++p2; }
            else               { im = mI[s3 * KNN + p3]; ++p3; }
            out_val[obase + k] = sqrtf(fmaxf(q2 + dm, 0.0f));
            out_idx[obase + k] = im;
        }
    }
}

extern "C" void kernel_launch(void* const* d_in, const int* in_sizes, int n_in,
                              void* d_out, int out_size, void* d_ws, size_t ws_size,
                              hipStream_t stream) {
    (void)in_sizes; (void)n_in; (void)out_size;
    const float* refp   = (const float*)d_in[0];
    const float* queryp = (const float*)d_in[1];
    float* out_val = (float*)d_out;
    int*   out_idx = (int*)d_out + (size_t)BATCH * NPTS * KNN;

    const size_t PLANE = (size_t)BATCH * NPTS * DIM * sizeof(unsigned short); // 8 MiB
    const size_t NORM  = (size_t)BATCH * NPTS * sizeof(float);                // 128 KiB
    const size_t WS_NEED = 4 * PLANE + 2 * NORM;                              // ~32.25 MiB

    dim3 grid(BATCH * (NPTS / QT));   // 512 workgroups
    dim3 block(THREADS);

    if (ws_size >= WS_NEED) {
        char* w = (char*)d_ws;
        unsigned short* refh = (unsigned short*)(w);
        unsigned short* refl = (unsigned short*)(w + PLANE);
        unsigned short* qh   = (unsigned short*)(w + 2 * PLANE);
        unsigned short* ql   = (unsigned short*)(w + 3 * PLANE);
        float*          ref2 = (float*)(w + 4 * PLANE);
        float*          q2   = (float*)(w + 4 * PLANE + NORM);

        const int rowsTotal = 2 * BATCH * NPTS;                 // one wave32 per row
        dim3 pgrid(rowsTotal / (THREADS / 32));                 // 8192 blocks
        hipLaunchKernelGGL(knn_prep_kernel, pgrid, block, 0, stream,
                           refp, queryp, refh, refl, qh, ql, ref2, q2);
        hipLaunchKernelGGL(knn_wmma_bf16_kernel, grid, block, 0, stream,
                           refh, refl, qh, ql, ref2, q2, out_val, out_idx);
    } else {
        hipLaunchKernelGGL(knn_wmma_f32_kernel, grid, block, 0, stream,
                           refp, queryp, out_val, out_idx);
    }
}
